// AttentionDecoder_14207751815248
// MI455X (gfx1250) — matrix operand order repaired and checked by reading, not compiled, not run on hardware
//
#include <hip/hip_runtime.h>
#include <hip/hip_bf16.h>
#include <stdint.h>
#include <stddef.h>

// ---------------- problem sizes ----------------
#define B_   128
#define S_   196
#define E_   2048
#define D_   512
#define A_   256
#define V_   10000
#define EM_  512
#define T_   20

typedef __bf16 bf16_t;
typedef __attribute__((ext_vector_type(8)))  __bf16 v8bf;
typedef __attribute__((ext_vector_type(16))) __bf16 v16bf;
typedef __attribute__((ext_vector_type(8)))  float  v8f;

static __device__ __forceinline__ v16bf frag_cat(v8bf a, v8bf b) {
  return __builtin_shufflevector(a, b, 0,1,2,3,4,5,6,7,8,9,10,11,12,13,14,15);
}

// ---------------- fp32 -> bf16 convert ----------------
__global__ void k_cvt_bf16(const float* __restrict__ in, bf16_t* __restrict__ out, int n) {
  int i = blockIdx.x * blockDim.x + threadIdx.x;
  if (i < n) out[i] = (bf16_t)in[i];
}

// ---------------- fused bias b_ih + b_hh ----------------
__global__ void k_add_bias(const float* __restrict__ a, const float* __restrict__ b,
                           float* __restrict__ o, int n) {
  int i = blockIdx.x * blockDim.x + threadIdx.x;
  if (i < n) o[i] = a[i] + b[i];
}

// ---------------- mean pool over S ----------------
__global__ void k_meanpool(const float* __restrict__ enc, float* __restrict__ mean) {
  int i = blockIdx.x * blockDim.x + threadIdx.x;     // b*E + e
  int b = i >> 11;                                   // /2048
  int e = i & (E_ - 1);
  const float* p = enc + (size_t)b * S_ * E_ + e;
  float acc = 0.f;
  for (int s = 0; s < S_; ++s) acc += p[(size_t)s * E_];
  mean[i] = acc * (1.0f / (float)S_);
}

// ---------------- embedding gather: [T,B,EM] fp32 ----------------
__global__ void k_gather_emb(const int* __restrict__ captions,
                             const float* __restrict__ table,
                             float* __restrict__ out) {
  int b = blockIdx.x, t = blockIdx.y;
  int tok = captions[b * T_ + t];
  const float* src = table + (size_t)tok * EM_;
  float* dst = out + ((size_t)t * B_ + b) * EM_;
  for (int e = threadIdx.x; e < EM_; e += blockDim.x) dst[e] = src[e];
}

// ---------------- WMMA bf16 GEMM: C[M,N] (+)= A_f32[M,K] * B_bf16[K,N] (+bias) --------
// flags: bit0 = accumulate into existing C, bit1 = add bias[n]
// 64x64 block tile, BK=64 double-buffered LDS, 4 waves (each a 32x32 quadrant),
// 8 x v_wmma_f32_16x16x32_bf16 per K-iteration, one barrier per iteration.
__launch_bounds__(128)
__global__ void k_gemm_wmma(const float* __restrict__ A, const bf16_t* __restrict__ Bm,
                            const float* __restrict__ bias, float* __restrict__ C,
                            int M, int N, int K, int lda, int ldb, int ldc, int flags) {
  __shared__ __align__(16) bf16_t As[2][64 * 64];   // [m][k], row stride 64
  __shared__ __align__(16) bf16_t Bs[2][64 * 64];   // [n][k] K-transposed, row stride 64

  const int tid  = threadIdx.x;
  const int lane = tid & 31;
  const int wave = tid >> 5;
  const int bm   = blockIdx.y * 64;
  const int bn   = blockIdx.x * 64;

  const int wm = (wave >> 1) * 32;        // wave quadrant row
  const int wn = (wave & 1) * 32;         // wave quadrant col

  const int fm   = lane & 15;             // row (A) / col (B) within 16-tile
  const int kb8  = (lane >> 4) << 3;      // A K-base: 0 or 8
  const int kb16 = (lane >> 4) << 3;      // B K-base: 0 or 8

  const v8f vzero = {0.f,0.f,0.f,0.f,0.f,0.f,0.f,0.f};
  v8f acc00 = vzero, acc01 = vzero, acc10 = vzero, acc11 = vzero;

  // per-thread staging coordinates: 64 rows x 64 K, 128 threads -> 32 elems each
  const int arow = tid >> 1;              // 0..63  (A row)
  const int acol = (tid & 1) * 32;        // 0 or 32 (A k-offset)
  const int bkr  = tid >> 1;              // 0..63  (B k-row)
  const int bnc  = (tid & 1) * 32;        // 0 or 32 (B n-offset)
  const int gn   = bn + bnc;

  float4 areg[8];                         // next-stage A (fp32)
  v8bf   breg[4];                         // next-stage B (bf16)

  auto load_global = [&](int k0) {
    const float* ap = A + (size_t)(bm + arow) * lda + k0 + acol;
    #pragma unroll
    for (int j = 0; j < 8; ++j) areg[j] = ((const float4*)ap)[j];
    const bf16_t* bp = Bm + (size_t)(k0 + bkr) * ldb + gn;
    if (gn + 32 <= N) {
      #pragma unroll
      for (int j = 0; j < 4; ++j) breg[j] = *(const v8bf*)(bp + 8 * j);
    } else {
      #pragma unroll
      for (int j = 0; j < 4; ++j)
        #pragma unroll
        for (int u = 0; u < 8; ++u)
          breg[j][u] = (gn + 8 * j + u < N) ? bp[8 * j + u] : (bf16_t)0.f;
    }
  };

  auto store_lds = [&](int buf) {
    #pragma unroll
    for (int j = 0; j < 4; ++j) {
      v8bf pk;
      float4 f0 = areg[2 * j], f1 = areg[2 * j + 1];
      pk[0]=(bf16_t)f0.x; pk[1]=(bf16_t)f0.y; pk[2]=(bf16_t)f0.z; pk[3]=(bf16_t)f0.w;
      pk[4]=(bf16_t)f1.x; pk[5]=(bf16_t)f1.y; pk[6]=(bf16_t)f1.z; pk[7]=(bf16_t)f1.w;
      *(v8bf*)&As[buf][arow * 64 + acol + 8 * j] = pk;
    }
    #pragma unroll
    for (int j = 0; j < 4; ++j)
      #pragma unroll
      for (int u = 0; u < 8; ++u)
        Bs[buf][(bnc + 8 * j + u) * 64 + bkr] = breg[j][u];
  };

  // prologue: stage K-block 0
  load_global(0);
  store_lds(0);
  __syncthreads();

  int cur = 0;
  for (int k0 = 0; k0 < K; k0 += 64) {
    const bool hasNext = (k0 + 64 < K);
    if (hasNext) {
      load_global(k0 + 64);                              // in flight during WMMAs
      if (k0 + 128 < K) {
        __builtin_prefetch(A + (size_t)(bm + arow) * lda + k0 + 128 + acol, 0, 1);
        __builtin_prefetch(Bm + (size_t)(k0 + 128 + bkr) * ldb + gn, 0, 1);
      }
    }

    // 2 k-steps x 4 WMMA from LDS[cur]
    #pragma unroll
    for (int ks = 0; ks < 2; ++ks) {
      const bf16_t* pa0 = &As[cur][(wm + fm) * 64 + ks * 32 + kb8];
      const bf16_t* pa1 = &As[cur][(wm + 16 + fm) * 64 + ks * 32 + kb8];
      v16bf a0 = frag_cat(*(const v8bf*)pa0, *(const v8bf*)(pa0 + 16));
      v16bf a1 = frag_cat(*(const v8bf*)pa1, *(const v8bf*)(pa1 + 16));
      const bf16_t* pb0 = &Bs[cur][(wn + fm) * 64 + ks * 32 + kb16];
      const bf16_t* pb1 = &Bs[cur][(wn + 16 + fm) * 64 + ks * 32 + kb16];
      v16bf b0 = frag_cat(*(const v8bf*)pb0, *(const v8bf*)(pb0 + 16));
      v16bf b1 = frag_cat(*(const v8bf*)pb1, *(const v8bf*)(pb1 + 16));

      acc00 = __builtin_amdgcn_wmma_f32_16x16x32_bf16(false, a0, false, b0, (short)0, acc00, false, false);
      acc01 = __builtin_amdgcn_wmma_f32_16x16x32_bf16(false, a0, false, b1, (short)0, acc01, false, false);
      acc10 = __builtin_amdgcn_wmma_f32_16x16x32_bf16(false, a1, false, b0, (short)0, acc10, false, false);
      acc11 = __builtin_amdgcn_wmma_f32_16x16x32_bf16(false, a1, false, b1, (short)0, acc11, false, false);
    }

    if (hasNext) store_lds(cur ^ 1);   // write other buffer (nobody reads it this iter)
    __syncthreads();                   // covers reads(cur) & writes(cur^1)
    cur ^= 1;
  }

  // ---- epilogue: C layout — VGPR r: lanes 0-15 M=r, lanes 16-31 M=r+8; N=lane&15 ----
  const int crow = (lane >> 4) * 8;
  const int ccol = lane & 15;
  v8f accs[2][2] = {{acc00, acc01}, {acc10, acc11}};
  #pragma unroll
  for (int i = 0; i < 2; ++i) {
    #pragma unroll
    for (int j = 0; j < 2; ++j) {
      int n = bn + wn + j * 16 + ccol;
      if (n < N) {
        float bv = (flags & 2) ? bias[n] : 0.f;
        #pragma unroll
        for (int r = 0; r < 8; ++r) {
          int m = bm + wm + i * 16 + crow + r;
          size_t idx = (size_t)m * ldc + n;
          float v = accs[i][j][r] + bv;
          if (flags & 1) v += C[idx];
          C[idx] = v;
        }
      }
    }
  }
}

// ---------------- fused Bahdanau attention: scores -> softmax -> context --------------
__launch_bounds__(256)
__global__ void k_attention(const float* __restrict__ encatt, const float* __restrict__ decatt,
                            const float* __restrict__ wfa, const float* __restrict__ bfa,
                            const float* __restrict__ enc, float* __restrict__ ctx) {
  const int b = blockIdx.x;
  const int tid = threadIdx.x;
  __shared__ float sDec[A_];
  __shared__ float sWfa[A_];
  __shared__ float sScore[256];
  __shared__ float sRed[8];
  __shared__ float sBcast;

  if (tid < A_) { sDec[tid] = decatt[b * A_ + tid]; sWfa[tid] = wfa[tid]; }
  __syncthreads();

  // scores[s] = sum_a relu(enc_att + dec_att) * W_fa  + b_fa
  for (int s = tid; s < S_; s += 256) {
    const float* ea = encatt + ((size_t)b * S_ + s) * A_;
    float acc = 0.f;
    for (int a = 0; a < A_; a += 4) {
      float4 e4 = *(const float4*)(ea + a);
      acc += fmaxf(e4.x + sDec[a + 0], 0.f) * sWfa[a + 0];
      acc += fmaxf(e4.y + sDec[a + 1], 0.f) * sWfa[a + 1];
      acc += fmaxf(e4.z + sDec[a + 2], 0.f) * sWfa[a + 2];
      acc += fmaxf(e4.w + sDec[a + 3], 0.f) * sWfa[a + 3];
    }
    sScore[s] = acc + bfa[0];
  }
  __syncthreads();

  // softmax over S (wave32 shuffle + LDS reduce)
  float m = -3.4e38f;
  for (int s = tid; s < S_; s += 256) m = fmaxf(m, sScore[s]);
  for (int off = 16; off > 0; off >>= 1) m = fmaxf(m, __shfl_down(m, off, 32));
  if ((tid & 31) == 0) sRed[tid >> 5] = m;
  __syncthreads();
  if (tid == 0) {
    float mm = sRed[0];
    for (int i = 1; i < 8; ++i) mm = fmaxf(mm, sRed[i]);
    sBcast = mm;
  }
  __syncthreads();
  m = sBcast;
  float sum = 0.f;
  for (int s = tid; s < S_; s += 256) { float e = __expf(sScore[s] - m); sScore[s] = e; sum += e; }
  for (int off = 16; off > 0; off >>= 1) sum += __shfl_down(sum, off, 32);
  if ((tid & 31) == 0) sRed[tid >> 5] = sum;
  __syncthreads();
  if (tid == 0) {
    float ss = 0.f;
    for (int i = 0; i < 8; ++i) ss += sRed[i];
    sBcast = 1.f / ss;
  }
  __syncthreads();
  float inv = sBcast;
  for (int s = tid; s < S_; s += 256) sScore[s] *= inv;
  __syncthreads();

  // context[b,e] = sum_s alpha[s] * enc[b,s,e]; thread owns 8 consecutive e
  const int e0 = tid * 8;
  float a0=0,a1=0,a2=0,a3=0,a4=0,a5=0,a6=0,a7=0;
  const float* base = enc + (size_t)b * S_ * E_ + e0;
  for (int s = 0; s < S_; ++s) {
    float al = sScore[s];
    float4 v0 = *(const float4*)(base + (size_t)s * E_);
    float4 v1 = *(const float4*)(base + (size_t)s * E_ + 4);
    a0 += al * v0.x; a1 += al * v0.y; a2 += al * v0.z; a3 += al * v0.w;
    a4 += al * v1.x; a5 += al * v1.y; a6 += al * v1.z; a7 += al * v1.w;
  }
  float* cp = ctx + (size_t)b * E_ + e0;
  float4 o0 = {a0, a1, a2, a3}, o1 = {a4, a5, a6, a7};
  *(float4*)cp = o0;
  *(float4*)(cp + 4) = o1;
}

// ---------------- LSTM pointwise ----------------
__global__ void k_lstm(const float* __restrict__ gates, float* __restrict__ h, float* __restrict__ c) {
  int i = blockIdx.x * blockDim.x + threadIdx.x;     // b*D + d
  if (i >= B_ * D_) return;
  int b = i / D_, d = i - b * D_;
  const float* g = gates + (size_t)b * 4 * D_;
  float gi = g[d], gf = g[D_ + d], gg = g[2 * D_ + d], go = g[3 * D_ + d];
  float si = 1.f / (1.f + __expf(-gi));
  float sf = 1.f / (1.f + __expf(-gf));
  float so = 1.f / (1.f + __expf(-go));
  float cn = sf * c[i] + si * tanhf(gg);
  float hn = so * tanhf(cn);
  c[i] = cn;
  h[i] = hn;
}

// ---------------- host launcher ----------------
extern "C" void kernel_launch(void* const* d_in, const int* in_sizes, int n_in,
                              void* d_out, int out_size, void* d_ws, size_t ws_size,
                              hipStream_t stream) {
  (void)in_sizes; (void)n_in; (void)out_size; (void)ws_size;
  const float* enc       = (const float*)d_in[0];
  const int*   captions  = (const int*)  d_in[1];
  const float* emb_table = (const float*)d_in[2];
  const float* W_ea = (const float*)d_in[3],  *b_ea = (const float*)d_in[4];
  const float* W_da = (const float*)d_in[5],  *b_da = (const float*)d_in[6];
  const float* W_fa = (const float*)d_in[7],  *b_fa = (const float*)d_in[8];
  const float* W_ih = (const float*)d_in[9],  *b_ih = (const float*)d_in[10];
  const float* W_hh = (const float*)d_in[11], *b_hh = (const float*)d_in[12];
  const float* W_cp = (const float*)d_in[13], *b_cp = (const float*)d_in[14];
  const float* W_hp = (const float*)d_in[15], *b_hp = (const float*)d_in[16];
  const float* W_op = (const float*)d_in[17], *b_op = (const float*)d_in[18];
  const float* W_inith = (const float*)d_in[19], *b_inith = (const float*)d_in[20];
  const float* W_initc = (const float*)d_in[21], *b_initc = (const float*)d_in[22];
  float* out = (float*)d_out;

  char* ws = (char*)d_ws;
  size_t off = 0;
  auto alloc = [&](size_t bytes) -> void* {
    off = (off + 255) & ~(size_t)255;
    void* p = ws + off;
    off += bytes;
    return p;
  };

  // bf16 weights (converted each call; deterministic/idempotent)
  bf16_t* Wea_b  = (bf16_t*)alloc((size_t)E_ * A_ * 2);
  bf16_t* Wda_b  = (bf16_t*)alloc((size_t)D_ * A_ * 2);
  bf16_t* Wih_b  = (bf16_t*)alloc((size_t)(EM_ + E_) * 4 * D_ * 2);
  bf16_t* Whh_b  = (bf16_t*)alloc((size_t)D_ * 4 * D_ * 2);
  bf16_t* Wcp_b  = (bf16_t*)alloc((size_t)E_ * EM_ * 2);
  bf16_t* Whp_b  = (bf16_t*)alloc((size_t)D_ * EM_ * 2);
  bf16_t* Wop_b  = (bf16_t*)alloc((size_t)EM_ * V_ * 2);
  bf16_t* Wih_b_ctx = Wih_b + (size_t)EM_ * 4 * D_;   // rows [EM, EM+E)
  bf16_t* Wini_h = (bf16_t*)alloc((size_t)E_ * D_ * 2);
  bf16_t* Wini_c = (bf16_t*)alloc((size_t)E_ * D_ * 2);

  // fp32 activations / scratch
  float* emb_f  = (float*)alloc((size_t)T_ * B_ * EM_ * 4);
  float* mean_f = (float*)alloc((size_t)B_ * E_ * 4);
  float* h_f    = (float*)alloc((size_t)B_ * D_ * 4);
  float* c_f    = (float*)alloc((size_t)B_ * D_ * 4);
  float* encatt = (float*)alloc((size_t)B_ * S_ * A_ * 4);
  float* decatt = (float*)alloc((size_t)B_ * A_ * 4);
  float* ctx    = (float*)alloc((size_t)B_ * E_ * 4);
  float* gates  = (float*)alloc((size_t)B_ * 4 * D_ * 4);
  float* feat   = (float*)alloc((size_t)B_ * EM_ * 4);
  float* bihh   = (float*)alloc((size_t)4 * D_ * 4);

  auto cvt = [&](const float* s, bf16_t* d, size_t n) {
    k_cvt_bf16<<<dim3((unsigned)((n + 255) / 256)), dim3(256), 0, stream>>>(s, d, (int)n);
  };
  cvt(W_ea, Wea_b, (size_t)E_ * A_);
  cvt(W_da, Wda_b, (size_t)D_ * A_);
  cvt(W_ih, Wih_b, (size_t)(EM_ + E_) * 4 * D_);
  cvt(W_hh, Whh_b, (size_t)D_ * 4 * D_);
  cvt(W_cp, Wcp_b, (size_t)E_ * EM_);
  cvt(W_hp, Whp_b, (size_t)D_ * EM_);
  cvt(W_op, Wop_b, (size_t)EM_ * V_);
  cvt(W_inith, Wini_h, (size_t)E_ * D_);
  cvt(W_initc, Wini_c, (size_t)E_ * D_);

  k_add_bias<<<dim3((4 * D_ + 255) / 256), dim3(256), 0, stream>>>(b_ih, b_hh, bihh, 4 * D_);
  k_meanpool<<<dim3((B_ * E_) / 256), dim3(256), 0, stream>>>(enc, mean_f);
  k_gather_emb<<<dim3(B_, T_), dim3(256), 0, stream>>>(captions, emb_table, emb_f);

  auto gemm = [&](const float* A, const bf16_t* Bm, const float* bias, float* C,
                  int M, int N, int K, int lda, int ldb, int ldc, int flags) {
    k_gemm_wmma<<<dim3((N + 63) / 64, (M + 63) / 64), dim3(128), 0, stream>>>(
        A, Bm, bias, C, M, N, K, lda, ldb, ldc, flags);
  };

  // init state + step-invariant encoder attention branch
  gemm(mean_f, Wini_h, b_inith, h_f, B_, D_, E_, E_, D_, D_, 2);
  gemm(mean_f, Wini_c, b_initc, c_f, B_, D_, E_, E_, D_, D_, 2);
  gemm(enc, Wea_b, b_ea, encatt, B_ * S_, A_, E_, E_, A_, A_, 2);

  for (int t = 0; t < T_; ++t) {
    // dec_att = h @ W_da + b_da
    gemm(h_f, Wda_b, b_da, decatt, B_, A_, D_, D_, A_, A_, 2);
    // alpha/context
    k_attention<<<dim3(B_), dim3(256), 0, stream>>>(encatt, decatt, W_fa, b_fa, enc, ctx);
    // gates = [emb; ctx] @ W_ih + h @ W_hh + (b_ih + b_hh)
    const float* xt = emb_f + (size_t)t * B_ * EM_;
    gemm(xt,  Wih_b,     bihh,    gates, B_, 4 * D_, EM_, EM_, 4 * D_, 4 * D_, 2);
    gemm(ctx, Wih_b_ctx, nullptr, gates, B_, 4 * D_, E_,  E_,  4 * D_, 4 * D_, 1);
    gemm(h_f, Whh_b,     nullptr, gates, B_, 4 * D_, D_,  D_,  4 * D_, 4 * D_, 1);
    // LSTM pointwise (updates h_f, c_f in place)
    k_lstm<<<dim3((B_ * D_ + 255) / 256), dim3(256), 0, stream>>>(gates, h_f, c_f);
    // feat = ctx @ W_cp + b_cp + h_new @ W_hp + b_hp
    gemm(ctx, Wcp_b, b_cp, feat, B_, EM_, E_, E_, EM_, EM_, 2);
    gemm(h_f, Whp_b, b_hp, feat, B_, EM_, D_, D_, EM_, EM_, 3);
    // logits -> out[b, t, :] with ldc = T*V
    gemm(feat, Wop_b, b_op, out + (size_t)t * V_, B_, V_, EM_, EM_, V_, T_ * V_, 2);
  }
}
